// TemporalFusionDecoder_13151189861026
// MI455X (gfx1250) — compile-verified, matrix operand order checked by reading, not run on hardware
//
#include <hip/hip_runtime.h>

// ---------------------------------------------------------------------------
// Types / WMMA helpers (gfx1250, wave32)
// ---------------------------------------------------------------------------
typedef _Float16 h16;
typedef __attribute__((ext_vector_type(16))) _Float16 v16h;
typedef __attribute__((ext_vector_type(8)))  _Float16 v8h;
typedef __attribute__((ext_vector_type(8)))  float    v8f;

__device__ __forceinline__ v8f wmma_f16(v16h a, v16h b, v8f c) {
  // D(16x16 f32) = A(16x32 f16) * B(32x16 f16) + C
  return __builtin_amdgcn_wmma_f32_16x16x32_f16(false, a, false, b, (short)0, c,
                                                false, false);
}

__device__ __forceinline__ v16h cat8(v8h a, v8h b) {
  v16h r;
#pragma unroll
  for (int i = 0; i < 8; ++i) { r[i] = a[i]; r[i + 8] = b[i]; }
  return r;
}

// A/B fragment loader from a row-major f16 buffer.
// lh = lane&15 selects the row (M for A / N for B); hi = lane>>4 selects the
// K sub-block: fragment = halfs [kb..kb+7] ++ [kb+16..kb+23] with kb = k0+hi*8.
__device__ __forceinline__ v16h load_frag(const h16* __restrict__ base,
                                          size_t stride, int row, int kb) {
  const h16* p = base + (size_t)row * stride + kb;
  return cat8(*(const v8h*)p, *(const v8h*)(p + 16));
}

// CDNA5 async tensor path: DMA 16B global -> LDS, tracked by ASYNCcnt.
// lds_off is the LDS byte offset (low 32 bits of a generic LDS pointer).
__device__ __forceinline__ void async_b128(unsigned lds_off,
                                           const h16* __restrict__ g) {
  asm volatile("global_load_async_to_lds_b128 %0, %1, off" ::"v"(lds_off),
               "v"(g)
               : "memory");
}
__device__ __forceinline__ void wait_async_5() {
  asm volatile("s_wait_asynccnt 0x5" ::: "memory");
}
__device__ __forceinline__ void wait_async_0() {
  asm volatile("s_wait_asynccnt 0x0" ::: "memory");
}

__device__ __forceinline__ float sigmoidf_(float x) {
  return 1.f / (1.f + __expf(-x));
}

// ---------------------------------------------------------------------------
// f32 -> f16 convert (zero-fills beyond srcN)
// ---------------------------------------------------------------------------
__global__ void __launch_bounds__(256) cvt_f32_f16_kernel(
    const float* __restrict__ src, h16* __restrict__ dst, int total, int srcN) {
  int i = blockIdx.x * 256 + threadIdx.x;
  if (i < total) dst[i] = (i < srcN) ? (h16)src[i] : (h16)0.0f;
}

// ---------------------------------------------------------------------------
// Generic WMMA GEMM:  C[M,N] = act( A[M,K] * W[N,K]^T + bias + rowAdd )
// Block: 256 threads (8 waves, 2x4), tile 64(M) x 256(N), K chunk 32,
// double-buffered LDS staged via global_load_async_to_lds_b128 (5 async ops
// per wave per chunk; OOB lanes branchlessly redirected to a zero page so the
// per-wave ASYNCcnt pipeline depth is exact).
// Wave tile 32x64 -> 8 WMMAs per chunk per wave.
// ---------------------------------------------------------------------------
#define LDT 40  // padded LDS row stride in halfs (80B: conflict-free, 16B-aligned)

__global__ void __launch_bounds__(256) gemm_f16_kernel(
    const h16* __restrict__ A, const h16* __restrict__ W,
    const float* __restrict__ bias, float* __restrict__ Cf,
    h16* __restrict__ Ch, const float* __restrict__ rowAdd, int rowAddDiv,
    const h16* __restrict__ zp, int M, int N, int K, int act) {
  __shared__ h16 As[2][64 * LDT];
  __shared__ h16 Ws[2][256 * LDT];
  const int tid = threadIdx.x;
  const int wave = tid >> 5, lane = tid & 31;
  const int lh = lane & 15, hi = lane >> 4;
  const int wy = wave >> 2, wx = wave & 3;  // 2 x 4 waves
  const int m0 = blockIdx.y * 64;
  const int n0 = blockIdx.x * 256;
  const int arow = tid >> 2;       // 0..63
  const int acol = (tid & 3) * 8;  // 0,8,16,24

  // per-thread staging sources (k-invariant row bounds -> uniform issue count)
  const h16* aRow = (m0 + arow < M) ? (A + (size_t)(m0 + arow) * K + acol) : zp;
  const bool aOk = (m0 + arow < M);
  const h16* wRow[4];
  bool wOk[4];
#pragma unroll
  for (int s = 0; s < 4; ++s) {
    int gn = n0 + arow + s * 64;
    wOk[s] = gn < N;
    wRow[s] = wOk[s] ? (W + (size_t)gn * K + acol) : zp;
  }

  auto stage = [&](int kc, int buf) {
    async_b128((unsigned)(uintptr_t)(&As[buf][arow * LDT + acol]),
               aOk ? aRow + kc : aRow);
#pragma unroll
    for (int s = 0; s < 4; ++s)
      async_b128((unsigned)(uintptr_t)(&Ws[buf][(arow + s * 64) * LDT + acol]),
                 wOk[s] ? wRow[s] + kc : wRow[s]);
  };

  v8f acc[2][4] = {};
  stage(0, 0);  // prologue

  for (int k0 = 0; k0 < K; k0 += 32) {
    const int cur = (k0 >> 5) & 1;
    __syncthreads();  // all waves done reading buf[cur^1]
    if (k0 + 32 < K) {
      stage(k0 + 32, cur ^ 1);
      wait_async_5();  // oldest 5 (current chunk) complete; next 5 in flight
    } else {
      wait_async_0();
    }
    __syncthreads();  // buf[cur] staged by every wave

    v16h af[2], bf[4];
#pragma unroll
    for (int i = 0; i < 2; ++i)
      af[i] = load_frag(As[cur], LDT, wy * 32 + i * 16 + lh, hi * 8);
#pragma unroll
    for (int j = 0; j < 4; ++j)
      bf[j] = load_frag(Ws[cur], LDT, wx * 64 + j * 16 + lh, hi * 8);
#pragma unroll
    for (int i = 0; i < 2; ++i)
#pragma unroll
      for (int j = 0; j < 4; ++j) acc[i][j] = wmma_f16(af[i], bf[j], acc[i][j]);
  }

  // epilogue
#pragma unroll
  for (int i = 0; i < 2; ++i)
#pragma unroll
    for (int j = 0; j < 4; ++j) {
      int col = n0 + wx * 64 + j * 16 + lh;
      if (col >= N) continue;
      float bv = bias ? bias[col] : 0.f;
#pragma unroll
      for (int r = 0; r < 8; ++r) {
        int row = m0 + wy * 32 + i * 16 + hi * 8 + r;
        if (row >= M) continue;
        float v = acc[i][j][r] + bv;
        if (rowAdd) v += rowAdd[(size_t)(row / rowAddDiv) * N + col];
        if (act == 1) v = v > 0.f ? v : __expf(v) - 1.f;  // ELU
        if (Cf) Cf[(size_t)row * N + col] = v;
        if (Ch) Ch[(size_t)row * N + col] = (h16)v;
      }
    }
}

// ---------------------------------------------------------------------------
// LSTM recurrence.  16 co-resident blocks; block b owns 32 hidden units.
// The 128 gate rows of Whh for those units stay resident in LDS for all
// timesteps; per-step h(t) is DMA'd into LDS via the async path.  Steps are
// separated by a device-scope monotonic atomic barrier across the 16 blocks.
// Dynamic LDS: 128*520*2 + 16*520*2 + 16*128*4 + 16*32*4 = 160000 bytes.
// ---------------------------------------------------------------------------
__global__ void __launch_bounds__(256) lstm_kernel(
    const float* __restrict__ xW,    // [16, Tl, 2048] rows m*Tl+t
    const h16* __restrict__ Whh,     // [2048, 512] f16
    h16* __restrict__ hbuf,          // 2 * 16*512 halfs (double buffer)
    const float* __restrict__ c_in,  // [16,512]
    float* __restrict__ c_out,       // [16,512]
    h16* __restrict__ hs16,          // [16, Ttot, 512] f16 hidden sequence
    int Tl, int t_off, int Ttot, unsigned* __restrict__ ctr) {
  extern __shared__ h16 smem[];
  h16* whh_s = smem;                      // 128 * 520 halfs
  h16* h_s = smem + 128 * 520;            // 16 * 520 halfs
  float* g_s = (float*)(h_s + 16 * 520);  // 16 * 128
  float* c_s = g_s + 16 * 128;            // 16 * 32

  const int tid = threadIdx.x;
  const int blk = blockIdx.x;  // 0..15
  const int jb = blk * 32;
  const int wave = tid >> 5, lane = tid & 31;
  const int lh = lane & 15, hi = lane >> 4;

  // preload Whh slice: lds row r = grp*32+jj  <-  global row grp*512+jb+jj
  {
    int r = tid >> 1, part = tid & 1;  // 128 rows, 2 threads/row
    int grp = r >> 5, jj = r & 31;
    const h16* src = Whh + (size_t)(grp * 512 + jb + jj) * 512 + part * 256;
    h16* dst = whh_s + r * 520 + part * 256;
#pragma unroll 4
    for (int k = 0; k < 256; k += 8)
      async_b128((unsigned)(uintptr_t)(dst + k), src + k);
  }
  for (int u = tid; u < 512; u += 256) {  // c init
    int m = u >> 5, jj = u & 31;
    c_s[u] = c_in[m * 512 + jb + jj];
  }
  wait_async_0();
  __syncthreads();

  for (int t = 0; t < Tl; ++t) {
    // DMA h(t) f16 16x512 into LDS (padded rows)
    const h16* hsrc = hbuf + (size_t)(t & 1) * (16 * 512);
    for (int u = tid * 8; u < 16 * 512; u += 256 * 8) {
      int m = u >> 9, k = u & 511;
      async_b128((unsigned)(uintptr_t)(h_s + m * 520 + k), hsrc + u);
    }
    wait_async_0();
    __syncthreads();

    // matmul: wave w computes gate columns [w*16, w*16+16) of this block
    v8f acc = {};
    for (int kc = 0; kc < 512; kc += 32) {
      v16h a = load_frag(h_s, 520, lh, kc + hi * 8);
      v16h b = load_frag(whh_s, 520, wave * 16 + lh, kc + hi * 8);
      acc = wmma_f16(a, b, acc);
    }
    {
      int c = wave * 16 + lh;           // 0..127 local gate column
      int grp = c >> 5, jjj = c & 31;
      int gcol = grp * 512 + jb + jjj;  // global gate column
#pragma unroll
      for (int r = 0; r < 8; ++r) {
        int m = hi * 8 + r;
        g_s[m * 128 + c] = acc[r] + xW[((size_t)m * Tl + t) * 2048 + gcol];
        if (t + 1 < Tl)
          __builtin_prefetch(&xW[((size_t)m * Tl + t + 1) * 2048 + gcol], 0, 1);
      }
    }
    __syncthreads();

    // gates + state update (512 (m,j) pairs per block)
    h16* hdst = hbuf + (size_t)((t + 1) & 1) * (16 * 512);
    for (int u = tid; u < 512; u += 256) {
      int m = u >> 5, jj = u & 31;
      float gi = g_s[m * 128 + jj];
      float gf = g_s[m * 128 + 32 + jj];
      float gg = g_s[m * 128 + 64 + jj];
      float go = g_s[m * 128 + 96 + jj];
      float c = sigmoidf_(gf) * c_s[u] + sigmoidf_(gi) * tanhf(gg);
      c_s[u] = c;
      float h = sigmoidf_(go) * tanhf(c);
      h16 hh = (h16)h;
      hdst[m * 512 + jb + jj] = hh;
      hs16[((size_t)m * Ttot + t_off + t) * 512 + jb + jj] = hh;
      if (t == Tl - 1) c_out[m * 512 + jb + jj] = c;
    }

    // device-scope barrier across the 16 blocks (monotonic counter)
    __threadfence();
    __syncthreads();
    if (tid == 0) {
      __hip_atomic_fetch_add(ctr, 1u, __ATOMIC_RELEASE,
                             __HIP_MEMORY_SCOPE_AGENT);
      unsigned target = 16u * (unsigned)(t + 1);
      while (__hip_atomic_load(ctr, __ATOMIC_ACQUIRE,
                               __HIP_MEMORY_SCOPE_AGENT) < target)
        __builtin_amdgcn_s_sleep(2);
    }
    __syncthreads();
  }
}

// ---------------------------------------------------------------------------
// GateAddNorm:  y = LN( a * sigmoid(bg) + skip ) * g + b      (D = 512)
// skip modes: 0 direct [r,512]; 1 concat(hist,fut) (T=768,S=512);
//             2 tail rows of a [*,768,512] buffer (t >= 512).
// ---------------------------------------------------------------------------
__global__ void __launch_bounds__(256) gate_ln_kernel(
    const float* __restrict__ a, const float* __restrict__ bg,
    const float* __restrict__ skipA, const float* __restrict__ skipB, int mode,
    const float* __restrict__ g, const float* __restrict__ beta,
    float* __restrict__ outF, h16* __restrict__ outH) {
  __shared__ float e_s[512];
  __shared__ float red[256];
  const int r = blockIdx.x, tid = threadIdx.x;

  for (int d = tid; d < 512; d += 256) {
    float av = a[(size_t)r * 512 + d];
    float bv = bg[(size_t)r * 512 + d];
    float sk;
    if (mode == 0) {
      sk = skipA[(size_t)r * 512 + d];
    } else if (mode == 1) {
      int bb = r / 768, t = r % 768;
      sk = (t < 512) ? skipA[((size_t)bb * 512 + t) * 512 + d]
                     : skipB[((size_t)bb * 256 + (t - 512)) * 512 + d];
    } else {
      int bb = r / 256, t = r % 256;
      sk = skipA[((size_t)bb * 768 + 512 + t) * 512 + d];
    }
    e_s[d] = av * sigmoidf_(bv) + sk;
  }
  __syncthreads();
  red[tid] = e_s[tid] + e_s[tid + 256];
  __syncthreads();
  for (int st = 128; st > 0; st >>= 1) {
    if (tid < st) red[tid] += red[tid + st];
    __syncthreads();
  }
  float mean = red[0] * (1.f / 512.f);
  __syncthreads();
  float d0 = e_s[tid] - mean, d1 = e_s[tid + 256] - mean;
  red[tid] = d0 * d0 + d1 * d1;
  __syncthreads();
  for (int st = 128; st > 0; st >>= 1) {
    if (tid < st) red[tid] += red[tid + st];
    __syncthreads();
  }
  float rstd = rsqrtf(red[0] * (1.f / 512.f) + 1e-5f);
  for (int d = tid; d < 512; d += 256) {
    float y = (e_s[d] - mean) * rstd * g[d] + beta[d];
    if (outF) outF[(size_t)r * 512 + d] = y;
    if (outH) outH[(size_t)r * 512 + d] = (h16)y;
  }
}

// ---------------------------------------------------------------------------
// Causal attention, queries restricted to the last P=256 positions.
// One wave per (qtile16, head, batch).  Flash-style online softmax.
// qkv f16 layout: [b*768+t, 1088] with q at 0, k at 512, v at 1024 (64-dim).
// ohead: [H=8][b*256+ql][64] f32, per-head (already scaled by 1/(H*l)).
// ---------------------------------------------------------------------------
__global__ void __launch_bounds__(32) attn_kernel(
    const h16* __restrict__ qkv, float* __restrict__ ohead) {
  __shared__ h16 p_s[16 * 32];
  __shared__ h16 vt[64 * LDT];
  const int qt = blockIdx.x, h = blockIdx.y, b = blockIdx.z;
  const int lane = threadIdx.x;
  const int lh = lane & 15, hi = lane >> 4;
  const size_t rowQ = (size_t)b * 768 + 512 + (size_t)qt * 16;

  v16h qf[2];
#pragma unroll
  for (int j = 0; j < 2; ++j)
    qf[j] = load_frag(qkv, 1088, (int)rowQ + lh, h * 64 + j * 32 + hi * 8);

  v8f o[4] = {};
  float rmax[8], lsum[8];
#pragma unroll
  for (int r = 0; r < 8; ++r) { rmax[r] = -__builtin_inff(); lsum[r] = 0.f; }

  const int nchunk = (528 + qt * 16 + 31) / 32;
  for (int c = 0; c < nchunk; ++c) {
    int kbase = c * 32;
    v8f s[2] = {};
#pragma unroll
    for (int j = 0; j < 2; ++j) {
      int krow = b * 768 + kbase + j * 16 + lh;
      v16h k0 = load_frag(qkv, 1088, krow, 512 + h * 64 + hi * 8);
      v16h k1 = load_frag(qkv, 1088, krow, 512 + h * 64 + 32 + hi * 8);
      s[j] = wmma_f16(qf[0], k0, s[j]);
      s[j] = wmma_f16(qf[1], k1, s[j]);
    }
    float mx[8];
#pragma unroll
    for (int r = 0; r < 8; ++r) {
      int qglob = 512 + qt * 16 + hi * 8 + r;
      float v0 = (kbase + lh <= qglob) ? s[0][r] * 0.125f : -__builtin_inff();
      float v1 =
          (kbase + 16 + lh <= qglob) ? s[1][r] * 0.125f : -__builtin_inff();
      s[0][r] = v0;
      s[1][r] = v1;
      mx[r] = fmaxf(v0, v1);
    }
#pragma unroll
    for (int m = 1; m < 16; m <<= 1)
#pragma unroll
      for (int r = 0; r < 8; ++r)
        mx[r] = fmaxf(mx[r], __shfl_xor(mx[r], m, 32));

    float corr[8], sm[8];
#pragma unroll
    for (int r = 0; r < 8; ++r) {
      float nm = fmaxf(rmax[r], mx[r]);
      corr[r] = __expf(rmax[r] - nm);
      rmax[r] = nm;
      float p0 = __expf(s[0][r] - nm);
      float p1 = __expf(s[1][r] - nm);
      p_s[(hi * 8 + r) * 32 + lh] = (h16)p0;
      p_s[(hi * 8 + r) * 32 + 16 + lh] = (h16)p1;
      sm[r] = p0 + p1;
    }
#pragma unroll
    for (int m = 1; m < 16; m <<= 1)
#pragma unroll
      for (int r = 0; r < 8; ++r) sm[r] += __shfl_xor(sm[r], m, 32);
#pragma unroll
    for (int r = 0; r < 8; ++r) lsum[r] = lsum[r] * corr[r] + sm[r];
#pragma unroll
    for (int nt = 0; nt < 4; ++nt)
#pragma unroll
      for (int r = 0; r < 8; ++r) o[nt][r] *= corr[r];

    {  // stage V chunk transposed: vt[d][key]; lane = key
      const h16* vsrc = qkv + ((size_t)b * 768 + kbase + lane) * 1088 + 1024;
#pragma unroll 8
      for (int d = 0; d < 64; ++d) vt[d * LDT + lane] = vsrc[d];
    }
    __syncthreads();
    v16h pf = load_frag(p_s, 32, lh, hi * 8);
#pragma unroll
    for (int nt = 0; nt < 4; ++nt) {
      v16h bf = load_frag(vt, LDT, nt * 16 + lh, hi * 8);
      o[nt] = wmma_f16(pf, bf, o[nt]);
    }
    __syncthreads();
  }

#pragma unroll
  for (int nt = 0; nt < 4; ++nt)
#pragma unroll
    for (int r = 0; r < 8; ++r) {
      int row = hi * 8 + r;
      ohead[(((size_t)h * 16 + b) * 256 + qt * 16 + row) * 64 + nt * 16 + lh] =
          o[nt][r] * (1.f / (8.f * lsum[r]));
    }
}

__global__ void __launch_bounds__(256) reduce_heads_kernel(
    const float* __restrict__ ohead, h16* __restrict__ o16, int n) {
  int i = blockIdx.x * 256 + threadIdx.x;
  if (i >= n) return;
  float s = 0.f;
#pragma unroll
  for (int h = 0; h < 8; ++h) s += ohead[(size_t)h * n + i];
  o16[i] = (h16)s;
}

// ---------------------------------------------------------------------------
// Host orchestration
// ---------------------------------------------------------------------------
static inline size_t align256(size_t x) { return (x + 255) & ~(size_t)255; }

extern "C" void kernel_launch(void* const* d_in, const int* in_sizes, int n_in,
                              void* d_out, int out_size, void* d_ws,
                              size_t ws_size, hipStream_t stream) {
  (void)in_sizes; (void)n_in; (void)out_size; (void)ws_size;
  const int D = 512, B = 16, S = 512, P = 256, T = 768;
  const int MH = B * S, MF = B * P, MT = B * T, MP = B * P;

  char* w = (char*)d_ws;
  size_t off = 0;
  auto alloc = [&](size_t bytes) -> char* {
    char* p = w + off;
    off = align256(off + bytes);
    return p;
  };

  // f16 weights
  h16* wih_h = (h16*)alloc(2048 * 512 * 2);
  h16* whh_h = (h16*)alloc(2048 * 512 * 2);
  h16* wih_f = (h16*)alloc(2048 * 512 * 2);
  h16* whh_f = (h16*)alloc(2048 * 512 * 2);
  h16* glf1 = (h16*)alloc(512 * 512 * 2);
  h16* glf2 = (h16*)alloc(512 * 512 * 2);
  h16* ea = (h16*)alloc(512 * 512 * 2);
  h16* ec = (h16*)alloc(512 * 512 * 2);
  h16* ei = (h16*)alloc(512 * 512 * 2);
  h16* egf1 = (h16*)alloc(512 * 512 * 2);
  h16* egf2 = (h16*)alloc(512 * 512 * 2);
  h16* qkvw = (h16*)alloc(1088 * 512 * 2);
  h16* aout = (h16*)alloc(512 * 64 * 2);
  h16* gaf1 = (h16*)alloc(512 * 512 * 2);
  h16* gaf2 = (h16*)alloc(512 * 512 * 2);
  h16* pa = (h16*)alloc(512 * 512 * 2);
  h16* pi = (h16*)alloc(512 * 512 * 2);
  h16* pgf1 = (h16*)alloc(512 * 512 * 2);
  h16* pgf2 = (h16*)alloc(512 * 512 * 2);
  h16* gff1 = (h16*)alloc(512 * 512 * 2);
  h16* gff2 = (h16*)alloc(512 * 512 * 2);
  h16* outw = (h16*)alloc(7 * 512 * 2);
  // small state
  h16* ce16 = (h16*)alloc(B * D * 2);
  h16* hbuf = (h16*)alloc(2 * B * D * 2);
  float* cT = (float*)alloc(B * D * 4);
  float* ceproj = (float*)alloc(B * D * 4);
  unsigned* ctr = (unsigned*)alloc(256);
  h16* zp = (h16*)alloc(256);  // zero page for OOB async-staging redirects
  // aliased big regions (re-used by lifetime)
  char* bigA = alloc((size_t)MH * 2048 * 4);      // xW_h -> attn-stage buffers
  char* bigB = alloc((size_t)MF * 2048 * 4);      // xW_f -> enrF + fin16
  char* actA = alloc((size_t)(MH + MF) * D * 2);  // hist16+fut16 -> x16
  char* hsR = alloc((size_t)MT * D * 2);          // hs16 -> x2_16
  float* abuf = (float*)alloc((size_t)MT * D * 4);
  float* bgbuf = (float*)alloc((size_t)MT * D * 4);
  float* tfF = (float*)alloc((size_t)MT * D * 4);
  h16* tf16 = (h16*)alloc((size_t)MT * D * 2);
  h16* enr16 = (h16*)alloc((size_t)MT * D * 2);

  float* xWh = (float*)bigA;
  float* xWf = (float*)bigB;
  h16* hist16 = (h16*)actA;
  h16* fut16 = hist16 + (size_t)MH * D;
  h16* x16 = (h16*)actA;
  h16* hs16 = (h16*)hsR;
  h16* x2_16 = (h16*)hsR;
  // bigA sub-layout after LSTMs
  h16* qkv16 = (h16*)bigA;                       // 12288x1088 f16
  float* ohead = (float*)(bigA + 26738688);      // 8x4096x64 f32
  h16* o16 = (h16*)(bigA + 26738688 + 8388608);  // 4096x64 f16
  h16* att16 = (h16*)(bigA + 26738688 + 8388608 + 524288);
  float* gaF = (float*)((char*)att16 + 4194304);
  h16* ga16 = (h16*)((char*)gaF + 8388608);
  h16* pwx16 = (h16*)((char*)ga16 + 4194304);
  h16* pwx2 = (h16*)((char*)pwx16 + 4194304);
  h16* po16 = (h16*)((char*)pwx2 + 4194304);
  // bigB sub-layout after LSTMs
  float* enrF = (float*)bigB;            // 12288x512 f32
  h16* fin16 = (h16*)(bigB + 25165824);  // 4096x512 f16

  auto cvt = [&](const void* src, h16* dst, int total, int srcN) {
    cvt_f32_f16_kernel<<<(total + 255) / 256, 256, 0, stream>>>(
        (const float*)src, dst, total, srcN);
  };
  auto gemm = [&](const h16* A, const h16* W, const float* bias, float* Cf,
                  h16* Ch, const float* rowAdd, int rdiv, int M, int N, int K,
                  int act) {
    dim3 g((N + 255) / 256, (M + 63) / 64);
    gemm_f16_kernel<<<g, 256, 0, stream>>>(A, W, bias, Cf, Ch, rowAdd, rdiv, zp,
                                           M, N, K, act);
  };
  auto gln = [&](const float* a, const float* bg, const float* sA,
                 const float* sB, int mode, const float* g, const float* be,
                 float* oF, h16* oH, int rows) {
    gate_ln_kernel<<<rows, 256, 0, stream>>>(a, bg, sA, sB, mode, g, be, oF,
                                             oH);
  };

  hipMemsetAsync(ctr, 0, 2 * sizeof(unsigned), stream);
  hipMemsetAsync(zp, 0, 256, stream);

  // ---- weight conversion ----
  cvt(d_in[5], wih_h, 2048 * 512, 2048 * 512);
  cvt(d_in[6], whh_h, 2048 * 512, 2048 * 512);
  cvt(d_in[8], wih_f, 2048 * 512, 2048 * 512);
  cvt(d_in[9], whh_f, 2048 * 512, 2048 * 512);
  cvt(d_in[11], glf1, 512 * 512, 512 * 512);
  cvt(d_in[13], glf2, 512 * 512, 512 * 512);
  cvt(d_in[17], ea, 512 * 512, 512 * 512);
  cvt(d_in[19], ec, 512 * 512, 512 * 512);
  cvt(d_in[21], ei, 512 * 512, 512 * 512);
  cvt(d_in[23], egf1, 512 * 512, 512 * 512);
  cvt(d_in[25], egf2, 512 * 512, 512 * 512);
  cvt(d_in[29], qkvw, 1088 * 512, 1088 * 512);
  cvt(d_in[30], aout, 512 * 64, 512 * 64);
  cvt(d_in[31], gaf1, 512 * 512, 512 * 512);
  cvt(d_in[33], gaf2, 512 * 512, 512 * 512);
  cvt(d_in[37], pa, 512 * 512, 512 * 512);
  cvt(d_in[39], pi, 512 * 512, 512 * 512);
  cvt(d_in[41], pgf1, 512 * 512, 512 * 512);
  cvt(d_in[43], pgf2, 512 * 512, 512 * 512);
  cvt(d_in[47], gff1, 512 * 512, 512 * 512);
  cvt(d_in[49], gff2, 512 * 512, 512 * 512);
  cvt(d_in[53], outw, 7 * 512, 7 * 512);
  // ---- activations / state conversion ----
  cvt(d_in[0], hist16, MH * D, MH * D);
  cvt(d_in[1], fut16, MF * D, MF * D);
  cvt(d_in[4], ce16, B * D, B * D);
  cvt(d_in[2], hbuf, B * D, B * D);  // h0 (= c_c arg) into buffer 0

  // ---- LSTM input projections: xW = x @ Wih^T + b ----
  gemm(hist16, wih_h, (const float*)d_in[7], xWh, nullptr, nullptr, 1, MH, 2048,
       D, 0);
  gemm(fut16, wih_f, (const float*)d_in[10], xWf, nullptr, nullptr, 1, MF, 2048,
       D, 0);

  // ---- LSTM recurrences (16 blocks, 160000B dynamic LDS each) ----
  hipFuncSetAttribute((const void*)lstm_kernel,
                      hipFuncAttributeMaxDynamicSharedMemorySize, 160000);
  lstm_kernel<<<16, 256, 160000, stream>>>(
      xWh, whh_h, hbuf, (const float*)d_in[3], cT, hs16, S, 0, T, ctr);
  lstm_kernel<<<16, 256, 160000, stream>>>(xWf, whh_f, hbuf, cT, cT, hs16, P, S,
                                           T, ctr + 1);

  // ---- gate_lstm: tf = GAN(concat(hist,fut)_h, skip = concat inputs) ----
  gemm(hs16, glf1, (const float*)d_in[12], abuf, nullptr, nullptr, 1, MT, D, D,
       0);
  gemm(hs16, glf2, (const float*)d_in[14], bgbuf, nullptr, nullptr, 1, MT, D, D,
       0);
  gln(abuf, bgbuf, (const float*)d_in[0], (const float*)d_in[1], 1,
      (const float*)d_in[15], (const float*)d_in[16], tfF, tf16, MT);

  // ---- ENR block ----
  gemm(ce16, ec, (const float*)d_in[20], ceproj, nullptr, nullptr, 1, B, D, D,
       0);
  gemm(tf16, ea, (const float*)d_in[18], nullptr, x16, ceproj, T, MT, D, D, 1);
  gemm(x16, ei, (const float*)d_in[22], nullptr, x2_16, nullptr, 1, MT, D, D,
       0);
  gemm(x2_16, egf1, (const float*)d_in[24], abuf, nullptr, nullptr, 1, MT, D, D,
       0);
  gemm(x2_16, egf2, (const float*)d_in[26], bgbuf, nullptr, nullptr, 1, MT, D,
       D, 0);
  gln(abuf, bgbuf, tfF, nullptr, 0, (const float*)d_in[27],
      (const float*)d_in[28], enrF, enr16, MT);

  // ---- attention (last P queries only) ----
  gemm(enr16, qkvw, nullptr, nullptr, qkv16, nullptr, 1, MT, 1088, D, 0);
  attn_kernel<<<dim3(16, 8, 16), 32, 0, stream>>>(qkv16, ohead);
  reduce_heads_kernel<<<(MP * 64 + 255) / 256, 256, 0, stream>>>(ohead, o16,
                                                                 MP * 64);
  gemm(o16, aout, nullptr, nullptr, att16, nullptr, 1, MP, D, 64, 0);
  gemm(att16, gaf1, (const float*)d_in[32], abuf, nullptr, nullptr, 1, MP, D, D,
       0);
  gemm(att16, gaf2, (const float*)d_in[34], bgbuf, nullptr, nullptr, 1, MP, D,
       D, 0);
  gln(abuf, bgbuf, enrF, nullptr, 2, (const float*)d_in[35],
      (const float*)d_in[36], gaF, ga16, MP);

  // ---- position-wise FF + gate ----
  gemm(ga16, pa, (const float*)d_in[38], nullptr, pwx16, nullptr, 1, MP, D, D,
       1);
  gemm(pwx16, pi, (const float*)d_in[40], nullptr, pwx2, nullptr, 1, MP, D, D,
       0);
  gemm(pwx2, pgf1, (const float*)d_in[42], abuf, nullptr, nullptr, 1, MP, D, D,
       0);
  gemm(pwx2, pgf2, (const float*)d_in[44], bgbuf, nullptr, nullptr, 1, MP, D, D,
       0);
  gln(abuf, bgbuf, gaF, nullptr, 0, (const float*)d_in[45],
      (const float*)d_in[46], nullptr, po16, MP);

  // ---- gate_final (skip = tf tail) ----
  gemm(po16, gff1, (const float*)d_in[48], abuf, nullptr, nullptr, 1, MP, D, D,
       0);
  gemm(po16, gff2, (const float*)d_in[50], bgbuf, nullptr, nullptr, 1, MP, D, D,
       0);
  gln(abuf, bgbuf, tfF, nullptr, 2, (const float*)d_in[51],
      (const float*)d_in[52], nullptr, fin16, MP);

  // ---- output projection: (B*P, 7) ----
  gemm(fin16, outw, (const float*)d_in[54], (float*)d_out, nullptr, nullptr, 1,
       MP, 7, D, 0);
}
